// graph_gru_gcn_19645180412754
// MI455X (gfx1250) — matrix operand order repaired and checked by reading, not compiled, not run on hardware
//
#include <hip/hip_runtime.h>
#include <hip/hip_bf16.h>
#include <math.h>

#define D 64

typedef __attribute__((ext_vector_type(16))) __bf16 v16bf;
typedef __attribute__((ext_vector_type(8)))  float  v8f;

__device__ __forceinline__ void fatomic_add(float* p, float v) {
  __hip_atomic_fetch_add(p, v, __ATOMIC_RELAXED, __HIP_MEMORY_SCOPE_AGENT);
}

// ---------------- utility kernels ----------------

__global__ void __launch_bounds__(256) k_fill(float* __restrict__ p, long long n, float v) {
  long long i = (long long)blockIdx.x * blockDim.x + threadIdx.x;
  if (i < n) p[i] = v;
}

__global__ void __launch_bounds__(256) k_deg(const int* __restrict__ row,
                                             float* __restrict__ deg, int nE) {
  int e = blockIdx.x * blockDim.x + threadIdx.x;
  if (e < nE) fatomic_add(&deg[row[e]], 1.0f);
}

__global__ void __launch_bounds__(256) k_dinv(float* __restrict__ d, int n) {
  int i = blockIdx.x * blockDim.x + threadIdx.x;
  if (i < n) {
    float v = d[i];
    d[i] = (v > 0.0f) ? rsqrtf(v) : 0.0f;
  }
}

// ---------------- weight pre-pack ----------------
// Split a 64x64 fp32 W into bf16 hi/lo B-fragments in the exact
// v_wmma_f32_16x16x32_bf16 B layout:
//   lane(lhalf=lane&15, hi=lane>>4): N = tn*16+lhalf, K(i) = k0 + (i>=8?16:0) + 8*hi + (i&7)
// Pack entry index: ((tn*2 + kt)*2 + part)*32 + lane, each entry = v16bf (32B).
// 512 entries (16 KB) per matrix; GEMM reads each fragment as two b128 loads.

__global__ void __launch_bounds__(256) k_packW(const float* __restrict__ W,
                                               v16bf* __restrict__ P) {
  const int t = threadIdx.x;          // 0..255
  const int lane = t & 31;
  const int kt   = (t >> 5) & 1;      // K-step (k0 = kt*32)
  const int tn   = t >> 6;            // column tile 0..3
  const int lhalf = lane & 15;
  const int hi    = lane >> 4;
  const float* p = W + (size_t)(kt * 32 + hi * 8) * D + tn * 16 + lhalf;
  v16bf bh, bl;
#pragma unroll
  for (int i = 0; i < 16; ++i) {
    float f = p[(size_t)(i + (i & 8)) * D];
    __bf16 b = (__bf16)f;
    bh[i] = b;
    bl[i] = (__bf16)(f - (float)b);
  }
  const size_t base = (size_t)((tn * 2 + kt) * 2) * 32 + lane;
  P[base]      = bh;   // part 0 = hi
  P[base + 32] = bl;   // part 1 = lo
}

// ---------------- WMMA GEMM: Out[N x 64] = X@Wx (+ H@Wh) (+ Out) ----------------
// One wave owns a 16x64 strip: A fragments (bf16 hi/lo split of fp32) loaded
// once with float4 loads, then swept across 4 pre-packed B column tiles.
// 3 WMMAs per (tile, K-step) give ~fp32 accuracy (hi*hi + hi*lo + lo*hi).

__device__ __forceinline__ void load_a_frag(const float* __restrict__ X, int row0,
                                            int lhalf, int hi, int k0,
                                            v16bf& ah, v16bf& al) {
  // A layout: M = row0+lhalf, K(i) = k0 + (i>=8?16:0) + 8*hi + (i&7)
  const float4* p = (const float4*)(X + (size_t)(row0 + lhalf) * D + k0 + hi * 8);
  float f[16];
  *(float4*)(f + 0)  = p[0];   // K offsets 0..3
  *(float4*)(f + 4)  = p[1];   // 4..7
  *(float4*)(f + 8)  = p[4];   // 16..19
  *(float4*)(f + 12) = p[5];   // 20..23
#pragma unroll
  for (int i = 0; i < 8; ++i) {
    __bf16 b0 = (__bf16)f[i];
    __bf16 b1 = (__bf16)f[8 + i];
    ah[i] = b0;
    ah[8 + i] = b1;
    al[i] = (__bf16)(f[i] - (float)b0);
    al[8 + i] = (__bf16)(f[8 + i] - (float)b1);
  }
}

__global__ void __launch_bounds__(256) gemm_strip(
    const float* __restrict__ X, const float* __restrict__ H,
    const v16bf* __restrict__ Px, const v16bf* __restrict__ Ph,
    float* __restrict__ Out, int nrows, int accumulate) {
  const int wave = (int)((blockIdx.x * blockDim.x + threadIdx.x) >> 5);
  const int lane = threadIdx.x & 31;
  if (wave * 16 >= nrows) return;     // wave-uniform: live waves keep EXEC all-1
  const int lhalf = lane & 15;
  const int hi    = lane >> 4;
  const int row0  = wave * 16;

  v8f acc[4] = {v8f{}, v8f{}, v8f{}, v8f{}};

#pragma unroll
  for (int src = 0; src < 2; ++src) {
    const float* A  = src ? H : X;
    const v16bf* P  = src ? Ph : Px;
    if (A == nullptr) continue;       // uniform branch
    v16bf ah[2], al[2];
#pragma unroll
    for (int kt = 0; kt < 2; ++kt)
      load_a_frag(A, row0, lhalf, hi, kt * 32, ah[kt], al[kt]);
#pragma unroll
    for (int tn = 0; tn < 4; ++tn) {
#pragma unroll
      for (int kt = 0; kt < 2; ++kt) {
        const size_t base = (size_t)((tn * 2 + kt) * 2) * 32 + lane;
        v16bf bh = P[base];
        v16bf bl = P[base + 32];
        acc[tn] = __builtin_amdgcn_wmma_f32_16x16x32_bf16(false, ah[kt], false, bh,
                                                          (short)0, acc[tn], false, false);
        acc[tn] = __builtin_amdgcn_wmma_f32_16x16x32_bf16(false, ah[kt], false, bl,
                                                          (short)0, acc[tn], false, false);
        acc[tn] = __builtin_amdgcn_wmma_f32_16x16x32_bf16(false, al[kt], false, bh,
                                                          (short)0, acc[tn], false, false);
      }
    }
  }

  // C layout: VGPR v -> row = row0 + v + 8*hi, col = tn*16 + lhalf
  const int r0 = row0 + hi * 8;
#pragma unroll
  for (int tn = 0; tn < 4; ++tn) {
    const int col = tn * 16 + lhalf;
#pragma unroll
    for (int v = 0; v < 8; ++v) {
      float* o = Out + (size_t)(r0 + v) * D + col;
      *o = accumulate ? (*o + acc[tn][v]) : acc[tn][v];
    }
  }
}

// ---------------- edge scatter (fused z+r pass, and h pass) ----------------

__global__ void __launch_bounds__(256) scatter_zr(
    const float* __restrict__ Yz, const float* __restrict__ Yr,
    const int* __restrict__ row, const int* __restrict__ col,
    const float* __restrict__ dinv,
    float* __restrict__ Az, float* __restrict__ Ar, int nE) {
  long long idx = (long long)blockIdx.x * blockDim.x + threadIdx.x;
  int e = (int)(idx >> 6);
  int d = (int)(idx & 63);
  if (e >= nE) return;
  int r = row[e], c = col[e];
  float nrm = dinv[r] * dinv[c];
  fatomic_add(&Az[(size_t)c * D + d], nrm * Yz[(size_t)r * D + d]);
  fatomic_add(&Ar[(size_t)c * D + d], nrm * Yr[(size_t)r * D + d]);
}

__global__ void __launch_bounds__(256) scatter_h(
    const float* __restrict__ Yh,
    const int* __restrict__ row, const int* __restrict__ col,
    const float* __restrict__ dinv, float* __restrict__ Ah, int nE) {
  long long idx = (long long)blockIdx.x * blockDim.x + threadIdx.x;
  int e = (int)(idx >> 6);
  int d = (int)(idx & 63);
  if (e >= nE) return;
  int r = row[e], c = col[e];
  float nrm = dinv[r] * dinv[c];
  fatomic_add(&Ah[(size_t)c * D + d], nrm * Yh[(size_t)r * D + d]);
}

// ---------------- per-node activations (self-loop folded in here) ----------------

__global__ void __launch_bounds__(256) act_zr(
    float* __restrict__ Az, float* __restrict__ Ar,
    const float* __restrict__ Yz, const float* __restrict__ Yr,
    const float* __restrict__ dinv,
    const float* __restrict__ bxz, const float* __restrict__ bhz,
    const float* __restrict__ bxr, const float* __restrict__ bhr,
    const float* __restrict__ hl, int nN) {
  long long idx = (long long)blockIdx.x * blockDim.x + threadIdx.x;
  int i = (int)(idx >> 6);
  int d = (int)(idx & 63);
  if (i >= nN) return;
  float di = dinv[i];
  float sn = 2.0f * di * di;          // self-loop norm (weight 2)
  size_t o = (size_t)i * D + d;
  float zpre = Az[o] + sn * Yz[o] + bxz[d] + bhz[d];
  float rpre = Ar[o] + sn * Yr[o] + bxr[d] + bhr[d];
  float z = 1.0f / (1.0f + __expf(-zpre));
  float r = 1.0f / (1.0f + __expf(-rpre));
  Az[o] = z;            // keep z for the final gate
  Ar[o] = r * hl[o];    // rh feeds the second GEMM
}

__global__ void __launch_bounds__(256) act_h(
    const float* __restrict__ Ah, const float* __restrict__ Yh,
    const float* __restrict__ dinv,
    const float* __restrict__ bxh, const float* __restrict__ bhh,
    const float* __restrict__ Z, const float* __restrict__ hl,
    float* __restrict__ out1, float* __restrict__ out2, int nN) {
  long long idx = (long long)blockIdx.x * blockDim.x + threadIdx.x;
  int i = (int)(idx >> 6);
  int d = (int)(idx & 63);
  if (i >= nN) return;
  float di = dinv[i];
  float sn = 2.0f * di * di;
  size_t o = (size_t)i * D + d;
  float pre = Ah[o] + sn * Yh[o] + bxh[d] + bhh[d];
  float ht = tanhf(pre);
  float z = Z[o];
  float ho = z * hl[o] + (1.0f - z) * ht;
  out1[o] = ho;   // h_out copy 1 (layer slice)
  out2[o] = ho;   // h_out copy 2 (tuple duplicates)
}

// ---------------- host launcher ----------------

extern "C" void kernel_launch(void* const* d_in, const int* in_sizes, int n_in,
                              void* d_out, int out_size, void* d_ws, size_t ws_size,
                              hipStream_t stream) {
  (void)n_in; (void)out_size; (void)ws_size;
  const float* inp = (const float*)d_in[0];
  const float* h   = (const float*)d_in[1];
  const int*   edg = (const int*)d_in[2];
  const float* Wxz = (const float*)d_in[3];
  const float* Whz = (const float*)d_in[4];
  const float* Wxr = (const float*)d_in[5];
  const float* Whr = (const float*)d_in[6];
  const float* Wxh = (const float*)d_in[7];
  const float* Whh = (const float*)d_in[8];
  const float* bxz = (const float*)d_in[9];
  const float* bhz = (const float*)d_in[10];
  const float* bxr = (const float*)d_in[11];
  const float* bhr = (const float*)d_in[12];
  const float* bxh = (const float*)d_in[13];
  const float* bhh = (const float*)d_in[14];
  float* out = (float*)d_out;

  const int N = in_sizes[0] / D;            // 100000 (multiple of 16)
  const int E = in_sizes[2] / 2;            // 1600000
  const int L = in_sizes[1] / in_sizes[0];  // 2

  // workspace (floats): dinv[N] | Yz,Yr,Yh[3*N*D] | Az,Ar[2*N*D] | 6 weight packs (16KB each)
  float* dinv = (float*)d_ws;
  float* Yz = dinv + N;
  float* Yr = Yz + (size_t)N * D;
  float* Yh = Yr + (size_t)N * D;
  float* Az = Yh + (size_t)N * D;   // reused as z after act_zr
  float* Ar = Az + (size_t)N * D;   // reused as rh, then as Ah
  v16bf* Pxz = (v16bf*)(Ar + (size_t)N * D);   // 32B-aligned (offset N*321*4 bytes)
  v16bf* Phz = Pxz + 512;
  v16bf* Pxr = Phz + 512;
  v16bf* Phr = Pxr + 512;
  v16bf* Pxh = Phr + 512;
  v16bf* Phh = Pxh + 512;

  const int TPB = 256;
  auto nblk = [](long long n) { return (unsigned)((n + 255) / 256); };

  // --- GCN normalization: deg = 2 (self loop) + in-edge count at row; dinv = deg^-1/2
  k_fill<<<nblk(N), TPB, 0, stream>>>(dinv, N, 2.0f);
  k_deg<<<nblk(E), TPB, 0, stream>>>(edg, dinv, E);
  k_dinv<<<nblk(N), TPB, 0, stream>>>(dinv, N);

  const int gemm_waves = N / 16;                        // one wave per 16x64 strip
  const unsigned gemm_blocks = (unsigned)((gemm_waves + 7) / 8);

  for (int l = 0; l < L; ++l) {
    const float* x_in = (l == 0) ? inp : (out + (size_t)(l - 1) * N * D);
    const float* hl = h + (size_t)l * N * D;
    const size_t wOff = (size_t)l * D * D;
    const size_t bOff = (size_t)l * D;

    // pre-pack this layer's weights into WMMA B-fragment layout (bf16 hi/lo)
    k_packW<<<1, TPB, 0, stream>>>(Wxz + wOff, Pxz);
    k_packW<<<1, TPB, 0, stream>>>(Whz + wOff, Phz);
    k_packW<<<1, TPB, 0, stream>>>(Wxr + wOff, Pxr);
    k_packW<<<1, TPB, 0, stream>>>(Whr + wOff, Phr);
    k_packW<<<1, TPB, 0, stream>>>(Wxh + wOff, Pxh);
    k_packW<<<1, TPB, 0, stream>>>(Whh + wOff, Phh);

    // fused pre-scatter GEMMs: Yz = x@Wxz + h@Whz ; Yr = x@Wxr + h@Whr ; Yh = x@Wxh
    gemm_strip<<<gemm_blocks, TPB, 0, stream>>>(x_in, hl, Pxz, Phz, Yz, N, 0);
    gemm_strip<<<gemm_blocks, TPB, 0, stream>>>(x_in, hl, Pxr, Phr, Yr, N, 0);
    gemm_strip<<<gemm_blocks, TPB, 0, stream>>>(x_in, nullptr, Pxh, nullptr, Yh, N, 0);

    // scatter pass 1 (z and r together)
    k_fill<<<nblk((long long)2 * N * D), TPB, 0, stream>>>(Az, (long long)2 * N * D, 0.0f);
    scatter_zr<<<nblk((long long)E * D), TPB, 0, stream>>>(Yz, Yr, edg, edg + E, dinv, Az, Ar, E);
    act_zr<<<nblk((long long)N * D), TPB, 0, stream>>>(Az, Ar, Yz, Yr, dinv,
                                                       bxz + bOff, bhz + bOff,
                                                       bxr + bOff, bhr + bOff, hl, N);

    // Yh += (r*h) @ Whh
    gemm_strip<<<gemm_blocks, TPB, 0, stream>>>(Ar, nullptr, Phh, nullptr, Yh, N, 1);

    // scatter pass 2 (h candidate), Ar reused as Ah
    k_fill<<<nblk((long long)N * D), TPB, 0, stream>>>(Ar, (long long)N * D, 0.0f);
    scatter_h<<<nblk((long long)E * D), TPB, 0, stream>>>(Yh, edg, edg + E, dinv, Ar, E);
    act_h<<<nblk((long long)N * D), TPB, 0, stream>>>(Ar, Yh, dinv,
                                                      bxh + bOff, bhh + bOff,
                                                      Az, hl,
                                                      out + (size_t)l * N * D,
                                                      out + (size_t)(L + l) * N * D, N);
  }
}
